// TBiDAFAttention_89120571392362
// MI455X (gfx1250) — matrix operand order repaired and checked by reading, not compiled, not run on hardware
//
#include <hip/hip_runtime.h>
#include <hip/hip_bf16.h>
#include <stdint.h>

// ---------------- problem constants ----------------
#define B_    16
#define LC    512
#define LQ    64
#define HID   192
#define HEADS 12
#define DH    1024              // INNER / HEADS
#define INNER 12288
#define SCALE 0.125f            // 64^-0.5
#define MASKV (-1e4f)
#define LN1E4_96 0.09594104554f // ln(10000)/96

// ---------------- WMMA fragment types ----------------
typedef __attribute__((ext_vector_type(16))) _Float16 v16h;
typedef __attribute__((ext_vector_type(8)))  _Float16 v8h;
typedef __attribute__((ext_vector_type(8)))  float    v8f;

__device__ __forceinline__ v8f wmma_f16(v16h a, v16h b, v8f c) {
  // v_wmma_f32_16x16x32_f16  D = A*B + C
  return __builtin_amdgcn_wmma_f32_16x16x32_f16(false, a, false, b, (short)0, c, false, false);
}

// A fragment: 16x32 f16, source row-major [M][K], ld in halves.
// lane: M = lane%16, hl = lane/16. elems 0..7 -> K = hl*8..+7 ; elems 8..15 -> K = 16+hl*8..+7
__device__ __forceinline__ v16h frag_a(const _Float16* p, int ld) {
  int lane = threadIdx.x & 31;
  int m = lane & 15, hl = lane >> 4;
  const _Float16* r = p + m * ld + hl * 8;
  v8h lo = *(const v8h*)r;
  v8h hi = *(const v8h*)(r + 16);
  return __builtin_shufflevector(lo, hi, 0,1,2,3,4,5,6,7,8,9,10,11,12,13,14,15);
}

// B fragment: 32x16 f16, source is B^T row-major [N][K], ld in halves.
// lane: N = lane%16, hl = lane/16. elems 0..7 -> K = hl*8..+7 ; elems 8..15 -> K = 16+hl*8..+7
__device__ __forceinline__ v16h frag_b(const _Float16* pT, int ld) {
  int lane = threadIdx.x & 31;
  int n = lane & 15, hl = lane >> 4;
  const _Float16* r = pT + n * ld + hl * 8;
  v8h lo = *(const v8h*)r;
  v8h hi = *(const v8h*)(r + 16);
  return __builtin_shufflevector(lo, hi, 0,1,2,3,4,5,6,7,8,9,10,11,12,13,14,15);
}

// ---------------- workspace layout (in halves) ----------------
#define OFF_QIN 0
#define OFF_KIN (OFF_QIN + (size_t)B_*LC*HID)
#define OFF_WQT (OFF_KIN + (size_t)B_*LQ*HID)
#define OFF_WKT (OFF_WQT + (size_t)HID*INNER)
#define OFF_WVT (OFF_WKT + (size_t)HID*INNER)
#define OFF_WOT (OFF_WVT + (size_t)HID*INNER)
#define OFF_KH  (OFF_WOT + (size_t)HID*INNER)
#define OFF_VHT (OFF_KH  + (size_t)B_*HEADS*LQ*DH)
#define OFF_AO  (OFF_VHT + (size_t)B_*HEADS*DH*LQ)

// ---------------- kernel 1: weight convert + transpose to f16 ----------------
__global__ void wconv_kernel(const float* __restrict__ Wq, const float* __restrict__ Wk,
                             const float* __restrict__ Wv, const float* __restrict__ Wo,
                             _Float16* __restrict__ WqT, _Float16* __restrict__ WkT,
                             _Float16* __restrict__ WvT, _Float16* __restrict__ WoT) {
  const int N = HID * INNER;
  for (int idx = blockIdx.x * blockDim.x + threadIdx.x; idx < N; idx += gridDim.x * blockDim.x) {
    int e = idx / INNER, f = idx % INNER;      // Wq/Wk/Wv row-major (HID, INNER)
    WqT[(size_t)f * HID + e] = (_Float16)Wq[idx];
    WkT[(size_t)f * HID + e] = (_Float16)Wk[idx];
    WvT[(size_t)f * HID + e] = (_Float16)Wv[idx];
    int f2 = idx / HID, e2 = idx % HID;        // Wo row-major (INNER, HID)
    WoT[(size_t)e2 * INNER + f2] = (_Float16)Wo[idx];
  }
}

// ---------------- kernel 2: prep (ScaleNorm + pos emb + c copy) ----------------
__device__ __forceinline__ float pos_emb(float t, int col) {
  if (col < 96) return sinf(t * expf(-(float)col * LN1E4_96));
  return cosf(t * expf(-(float)(col - 96) * LN1E4_96));
}

__global__ void prep_kernel(const float* __restrict__ c, const float* __restrict__ q,
                            const float* __restrict__ g,
                            _Float16* __restrict__ qin, _Float16* __restrict__ kin,
                            float* __restrict__ out) {
  int wave = threadIdx.x >> 5, lane = threadIdx.x & 31;
  int gb = blockIdx.x;
  if (gb < (B_ * LC / 8)) {                      // q_input rows (ScaleNorm'd c + emb)
    int row = gb * 8 + wave;                     // 0..8191
    int i = row & (LC - 1);
    const float* cr = c + (size_t)row * HID;
    float vals[6];
    float ss = 0.f;
#pragma unroll
    for (int k = 0; k < 6; ++k) { float v = cr[lane + 32 * k]; vals[k] = v; ss += v * v; }
#pragma unroll
    for (int off = 16; off; off >>= 1) ss += __shfl_xor(ss, off, 32);
    float nrm = fmaxf(sqrtf(ss), 1e-5f);
    float sc = g[0] / nrm;
    float t = (float)i + (float)(LQ - LC);       // offset = lq - lc
#pragma unroll
    for (int k = 0; k < 6; ++k) {
      int col = lane + 32 * k;
      qin[(size_t)row * HID + col] = (_Float16)(vals[k] * sc + pos_emb(t, col));
      out[(size_t)row * (2 * HID) + col] = vals[k];   // concat first half = c
    }
  } else {                                       // k_input rows (raw q + emb)
    int row = (gb - (B_ * LC / 8)) * 8 + wave;   // 0..1023
    int j = row & (LQ - 1);
    const float* qr = q + (size_t)row * HID;
    float t = (float)j;
#pragma unroll
    for (int k = 0; k < 6; ++k) {
      int col = lane + 32 * k;
      kin[(size_t)row * HID + col] = (_Float16)(qr[col] + pos_emb(t, col));
    }
  }
}

// ---------------- kernel 3: K/V projections per (b,h) ----------------
__global__ void kv_kernel(const _Float16* __restrict__ kin,
                          const _Float16* __restrict__ WkT, const _Float16* __restrict__ WvT,
                          _Float16* __restrict__ Kh, _Float16* __restrict__ VhT) {
  __shared__ _Float16 kinb[LQ * HID];
  int gid = blockIdx.x;                 // b*HEADS + h
  int b = gid / HEADS, h = gid % HEADS;
  const _Float16* src = kin + (size_t)b * LQ * HID;
  for (int i = threadIdx.x; i < LQ * HID / 8; i += blockDim.x)
    ((v8h*)kinb)[i] = ((const v8h*)src)[i];
  __syncthreads();

  int wave = threadIdx.x >> 5, lane = threadIdx.x & 31;
  int nlane = lane & 15, hl = lane >> 4;
  _Float16* Khb = Kh  + (size_t)gid * LQ * DH;   // row-major [j][d]
  _Float16* Vtb = VhT + (size_t)gid * DH * LQ;   // row-major [d][j] (transposed V)

  for (int idx = wave; idx < 256; idx += 8) {    // 4 x 64 tiles of 16x16
    int mi = idx & 3, dj = idx >> 2;
    v8f ak = {}, av = {};
    const _Float16* wk = WkT + (size_t)(h * DH + dj * 16) * HID;
    const _Float16* wv = WvT + (size_t)(h * DH + dj * 16) * HID;
#pragma unroll
    for (int kk = 0; kk < 6; ++kk) {
      v16h a  = frag_a(kinb + mi * 16 * HID + kk * 32, HID);
      ak = wmma_f16(a, frag_b(wk + kk * 32, HID), ak);
      av = wmma_f16(a, frag_b(wv + kk * 32, HID), av);
    }
#pragma unroll
    for (int r = 0; r < 8; ++r)
      Khb[(size_t)(mi * 16 + r + 8 * hl) * DH + dj * 16 + nlane] = (_Float16)ak[r];
    v8h pk;                                      // transposed store: contiguous in j
#pragma unroll
    for (int r = 0; r < 8; ++r) pk[r] = (_Float16)av[r];
    *(v8h*)(Vtb + (size_t)(dj * 16 + nlane) * LQ + mi * 16 + 8 * hl) = pk;
  }
}

// ---------------- kernel 4: fused Q-proj + dots + entmax1.5 + attn@V ----------------
__global__ void attn_kernel(const _Float16* __restrict__ qin, const _Float16* __restrict__ WqT,
                            const _Float16* __restrict__ Kh,  const _Float16* __restrict__ VhT,
                            const uint8_t* __restrict__ c_mask, const uint8_t* __restrict__ q_mask,
                            _Float16* __restrict__ AO) {
  __shared__ _Float16 qinb[64 * HID];   // 24 KB
  __shared__ _Float16 qch[64 * 32];     //  4 KB  (Q chunk, f16)
  __shared__ float    dotsL[64 * 64];   // 16 KB
  __shared__ _Float16 attnL[64 * 64];   //  8 KB

  int gid = blockIdx.x;                 // b*96 + h*8 + blk
  int blk = gid & 7, h = (gid >> 3) % HEADS, b = gid / (HEADS * 8);
  int row0 = b * LC + blk * 64;

  const _Float16* src = qin + (size_t)row0 * HID;
  for (int i = threadIdx.x; i < 64 * HID / 8; i += blockDim.x)
    ((v8h*)qinb)[i] = ((const v8h*)src)[i];
  __syncthreads();

  int wave = threadIdx.x >> 5, lane = threadIdx.x & 31;
  int nlane = lane & 15, hl = lane >> 4;
  const _Float16* Khb = Kh  + (size_t)(b * HEADS + h) * LQ * DH;
  const _Float16* Vtb = VhT + (size_t)(b * HEADS + h) * DH * LQ;

  int mi = wave & 3, ni2 = wave >> 2;            // this wave's Q-chunk tile
  int t0 = wave, t1 = wave + 8;                  // this wave's two dots tiles
  int m0 = t0 & 3, j0 = t0 >> 2;
  int m1 = t1 & 3, j1 = t1 >> 2;
  v8f acc0 = {}, acc1 = {};

  for (int kc = 0; kc < DH / 32; ++kc) {
    // project one 64x32 Q chunk into LDS (8 tiles, one per wave)
    v8f aq = {};
    const _Float16* wq = WqT + (size_t)(h * DH + kc * 32 + ni2 * 16) * HID;
#pragma unroll
    for (int kk = 0; kk < 6; ++kk)
      aq = wmma_f16(frag_a(qinb + mi * 16 * HID + kk * 32, HID),
                    frag_b(wq + kk * 32, HID), aq);
#pragma unroll
    for (int r = 0; r < 8; ++r)
      qch[(mi * 16 + r + 8 * hl) * 32 + ni2 * 16 + nlane] = (_Float16)aq[r];
    __syncthreads();
    // accumulate dots tiles against Kh
    acc0 = wmma_f16(frag_a(qch + m0 * 16 * 32, 32),
                    frag_b(Khb + (size_t)(j0 * 16) * DH + kc * 32, DH), acc0);
    acc1 = wmma_f16(frag_a(qch + m1 * 16 * 32, 32),
                    frag_b(Khb + (size_t)(j1 * 16) * DH + kc * 32, DH), acc1);
    __syncthreads();
  }
#pragma unroll
  for (int r = 0; r < 8; ++r) {
    dotsL[(m0 * 16 + r + 8 * hl) * 64 + j0 * 16 + nlane] = acc0[r];
    dotsL[(m1 * 16 + r + 8 * hl) * 64 + j1 * 16 + nlane] = acc1[r];
  }
  __syncthreads();

  // ---- exact entmax-1.5 via bisection on tau (g(tau)=sum relu(x-tau)^2 is monotone) ----
  uint8_t qm0 = q_mask[b * LQ + lane];
  uint8_t qm1 = q_mask[b * LQ + lane + 32];
  for (int rr = 0; rr < 8; ++rr) {
    int row = wave * 8 + rr;
    uint8_t cm = c_mask[row0 + row];
    float x0 = (cm && qm0) ? dotsL[row * 64 + lane] * SCALE : MASKV;
    float x1 = (cm && qm1) ? dotsL[row * 64 + lane + 32] * SCALE : MASKV;
    x0 *= 0.5f; x1 *= 0.5f;
    float mx = fmaxf(x0, x1);
#pragma unroll
    for (int off = 16; off; off >>= 1) mx = fmaxf(mx, __shfl_xor(mx, off, 32));
    x0 -= mx; x1 -= mx;
    float lo = -1.0f, hi = 0.0f;                 // g(-1)>=1 >= g(0)
    for (int it = 0; it < 30; ++it) {
      float mid = 0.5f * (lo + hi);
      float r0 = fmaxf(x0 - mid, 0.f), r1 = fmaxf(x1 - mid, 0.f);
      float gg = r0 * r0 + r1 * r1;
#pragma unroll
      for (int off = 16; off; off >>= 1) gg += __shfl_xor(gg, off, 32);
      if (gg > 1.0f) lo = mid; else hi = mid;
    }
    float tau = 0.5f * (lo + hi);
    float p0 = fmaxf(x0 - tau, 0.f); p0 *= p0;
    float p1 = fmaxf(x1 - tau, 0.f); p1 *= p1;
    attnL[row * 64 + lane]      = (_Float16)p0;
    attnL[row * 64 + lane + 32] = (_Float16)p1;
  }
  __syncthreads();

  // ---- out = attn @ V  (B-fragments from transposed V, contiguous loads) ----
  _Float16* AOb = AO + (size_t)row0 * INNER + h * DH;
  for (int idx = wave; idx < 256; idx += 8) {    // 4 x 64 tiles
    int mo = idx & 3, dj = idx >> 2;
    v8f acc = {};
#pragma unroll
    for (int kch = 0; kch < 2; ++kch)
      acc = wmma_f16(frag_a(attnL + mo * 16 * 64 + kch * 32, 64),
                     frag_b(Vtb + (size_t)(dj * 16) * LQ + kch * 32, LQ), acc);
#pragma unroll
    for (int r = 0; r < 8; ++r)
      AOb[(size_t)(mo * 16 + r + 8 * hl) * INNER + dj * 16 + nlane] = (_Float16)acc[r];
  }
}

// ---------------- kernel 5: output projection + bias + residual ----------------
__global__ void final_kernel(const _Float16* __restrict__ AO, const _Float16* __restrict__ WoT,
                             const float* __restrict__ bo, const float* __restrict__ c,
                             float* __restrict__ out) {
  int gid = blockIdx.x;                 // 256 blocks: b in [0,16), 16 row-blocks of 32
  int b = gid >> 4, rb = (gid & 15) * 32;
  int wave = threadIdx.x >> 5, lane = threadIdx.x & 31;
  int nlane = lane & 15, hl = lane >> 4;
  const _Float16* Arow = AO + (size_t)(b * LC + rb) * INNER;
  for (int idx = wave; idx < 24; idx += 8) {     // 2 x 12 tiles of 16x16
    int mi = idx & 1, nj = idx >> 1;
    v8f acc = {};
    for (int ch = 0; ch < INNER / 32; ++ch)
      acc = wmma_f16(frag_a(Arow + (size_t)(mi * 16) * INNER + ch * 32, INNER),
                     frag_b(WoT + (size_t)(nj * 16) * INNER + ch * 32, INNER), acc);
    int col = nj * 16 + nlane;
    float bv = bo[col];
#pragma unroll
    for (int r = 0; r < 8; ++r) {
      size_t rowidx = (size_t)(b * LC + rb + mi * 16 + r + 8 * hl);
      out[rowidx * (2 * HID) + HID + col] = acc[r] + bv + c[rowidx * HID + col];
    }
  }
}

// ---------------- launcher ----------------
extern "C" void kernel_launch(void* const* d_in, const int* in_sizes, int n_in,
                              void* d_out, int out_size, void* d_ws, size_t ws_size,
                              hipStream_t stream) {
  const float*   c      = (const float*)d_in[0];
  const float*   q      = (const float*)d_in[1];
  const uint8_t* c_mask = (const uint8_t*)d_in[2];
  const uint8_t* q_mask = (const uint8_t*)d_in[3];
  const float*   g      = (const float*)d_in[4];
  const float*   Wq     = (const float*)d_in[5];
  const float*   Wk     = (const float*)d_in[6];
  const float*   Wv     = (const float*)d_in[7];
  const float*   Wo     = (const float*)d_in[8];
  const float*   bo     = (const float*)d_in[9];
  float* out = (float*)d_out;

  _Float16* ws  = (_Float16*)d_ws;
  _Float16* qin = ws + OFF_QIN;
  _Float16* kin = ws + OFF_KIN;
  _Float16* WqT = ws + OFF_WQT;
  _Float16* WkT = ws + OFF_WKT;
  _Float16* WvT = ws + OFF_WVT;
  _Float16* WoT = ws + OFF_WOT;
  _Float16* Kh  = ws + OFF_KH;
  _Float16* VhT = ws + OFF_VHT;
  _Float16* AO  = ws + OFF_AO;

  wconv_kernel<<<2048, 256, 0, stream>>>(Wq, Wk, Wv, Wo, WqT, WkT, WvT, WoT);
  prep_kernel<<<(B_ * LC / 8) + (B_ * LQ / 8), 256, 0, stream>>>(c, q, g, qin, kin, out);
  kv_kernel<<<B_ * HEADS, 256, 0, stream>>>(kin, WkT, WvT, Kh, VhT);
  attn_kernel<<<B_ * HEADS * 8, 256, 0, stream>>>(qin, WqT, Kh, VhT, c_mask, q_mask, AO);
  final_kernel<<<256, 256, 0, stream>>>(AO, WoT, bo, c, out);
}
